// PivotalAttention_77386720739548
// MI455X (gfx1250) — matrix operand order repaired and checked by reading, not compiled, not run on hardware
//
#include <hip/hip_runtime.h>

typedef _Float16 half_t;
typedef __attribute__((ext_vector_type(16))) _Float16 v16h;
typedef __attribute__((ext_vector_type(8)))  float    v8f;

#define WMMA_F16(a, b, c) \
  __builtin_amdgcn_wmma_f32_16x16x32_f16(false, (a), false, (b), (short)0, (c), false, false)

// Problem constants
#define BB     8
#define HH     12
#define LL     500
#define DD     152
#define BH     (BB*HH)          // 96
#define NTOP   70
#define SAMPK  70
#define LLP    512              // padded L
#define DP     160              // padded D (5 k-steps of 32)
#define SBH    (LLP*DP)         // halfs per (b,h) panel (81920)
#define KSTEPS_D 5
#define NTILES_D 10

union F16Frag { v16h v; uint4 q[2]; };
union H2 { unsigned u; _Float16 h[2]; };
union H8 { uint4 q; _Float16 h[8]; };

__device__ __forceinline__ float neg_inf() { return -__builtin_inff(); }

// ---------------- WMMA fragment loaders: unconditional packed b128 reads ----------------
// A (16x32 f16): lane row = lane&15, kbase=(lane>=16)*8; VGPR v<4 -> K=kbase+2v(+1), v>=4 -> +16.
//   => dwords [base..base+3] and [base+8..base+11]; base = row*ld + k0 + kbase (16B aligned).
__device__ __forceinline__ v16h frag_a(const half_t* p, int ldh, int row, int k0, int lane) {
  const uint4* q = (const uint4*)(p + row * ldh + k0 + ((lane >> 4) << 3));
  F16Frag f; f.q[0] = q[0]; f.q[1] = q[2];
  return f.v;
}
// B (32x16 f16) from memory laid out [col][k]: lane col = lane&15, kbase=(lane>=16)*8; same K order as A;
//   => dwords [base..base+3] and [base+8..base+11].
__device__ __forceinline__ v16h frag_bt(const half_t* p, int ldh, int k0, int lane) {
  const uint4* q = (const uint4*)(p + (lane & 15) * ldh + k0 + ((lane >> 4) << 3));
  F16Frag f; f.q[0] = q[0]; f.q[1] = q[2];
  return f.v;
}

// ---------------- Prep: X -> padded f16 panels Xhp[bh][512][160] ----------------
__global__ __launch_bounds__(256)
void prep_x_kernel(const float* __restrict__ X, half_t* __restrict__ Xhp) {
  int id = blockIdx.x * 256 + threadIdx.x;           // one dword (f16 pair)
  if (id >= BH * LLP * (DP / 2)) return;
  int bh = id / (LLP * (DP / 2));
  int rem = id - bh * (LLP * (DP / 2));
  int n = rem / (DP / 2);
  int pr = rem - n * (DP / 2);
  int k = pr * 2;
  bool ok = (n < LL) & (k < DD);
  size_t off = (size_t)(bh * LL + (ok ? n : 0)) * DD + (ok ? k : 0);
  float2 f = *(const float2*)(X + off);
  H2 t; t.h[0] = (_Float16)(ok ? f.x : 0.f); t.h[1] = (_Float16)(ok ? f.y : 0.f);
  ((unsigned*)Xhp)[id] = t.u;
}

// ---------------- Prep: Wq/Wk/Wv -> padded f16 [160][160] (row-major == B layout) ----------
__global__ __launch_bounds__(256)
void prep_w_kernel(const float* __restrict__ Wq, const float* __restrict__ Wk,
                   const float* __restrict__ Wv, half_t* __restrict__ Wp) {
  int id = blockIdx.x * 256 + threadIdx.x;           // 3 * 160 * 80 dwords
  if (id >= 3 * DP * (DP / 2)) return;
  int m = id / (DP * (DP / 2));
  int rem = id - m * (DP * (DP / 2));
  int r = rem / (DP / 2);
  int pr = rem - r * (DP / 2);
  int k = pr * 2;
  const float* W = (m == 0) ? Wq : (m == 1) ? Wk : Wv;
  bool ok = (r < DD) & (k < DD);
  float2 f = *(const float2*)(W + (size_t)(ok ? r : 0) * DD + (ok ? k : 0));
  H2 t; t.h[0] = (_Float16)(ok ? f.x : 0.f); t.h[1] = (_Float16)(ok ? f.y : 0.f);
  ((unsigned*)Wp)[id] = t.u;
}

// ---------------- Kernel: fused QKV projection + bias + ReLU (pure WMMA) ----------------
// One block per 16-row tile; A fragments for all 5 k-steps held in registers and
// reused across all 10 column tiles (150 WMMAs per wave).
__global__ __launch_bounds__(32)
void qkv_kernel(const half_t* __restrict__ Xhp, const half_t* __restrict__ Wp,
                const float* __restrict__ bq, const float* __restrict__ bk,
                const float* __restrict__ bv,
                half_t* __restrict__ Qhp, half_t* __restrict__ Khp,
                half_t* __restrict__ Vhp) {
  const int lane = threadIdx.x;
  const int m0 = blockIdx.x * 16;          // [0, BH*512)
  const int arow = m0 + (lane & 15);
  const int rbase = m0 + ((lane >> 4) << 3);

  v16h a[KSTEPS_D];
#pragma unroll
  for (int ks = 0; ks < KSTEPS_D; ++ks) a[ks] = frag_a(Xhp, DP, arow, ks * 32, lane);

  for (int nt = 0; nt < NTILES_D; ++nt) {
    const int n0 = nt * 16;
    const half_t* Bq = Wp + 0 * DP * DP + n0 * DP;
    const half_t* Bk = Wp + 1 * DP * DP + n0 * DP;
    const half_t* Bv = Wp + 2 * DP * DP + n0 * DP;
    v8f cq = {}, ck = {}, cv = {};
#pragma unroll
    for (int ks = 0; ks < KSTEPS_D; ++ks) {
      const int k0 = ks * 32;
      cq = WMMA_F16(a[ks], frag_bt(Bq, DP, k0, lane), cq);
      ck = WMMA_F16(a[ks], frag_bt(Bk, DP, k0, lane), ck);
      cv = WMMA_F16(a[ks], frag_bt(Bv, DP, k0, lane), cv);
    }
    const int col = n0 + (lane & 15);
    const bool colv = col < DD;
    const int bcol = colv ? col : 0;
    const float biq = bq[bcol], bik = bk[bcol], biv = bv[bcol];
#pragma unroll
    for (int r = 0; r < 8; ++r) {
      size_t o = (size_t)(rbase + r) * DP + col;
      float q = colv ? fmaxf(cq[r] + biq, 0.f) : 0.f;
      float k = colv ? fmaxf(ck[r] + bik, 0.f) : 0.f;
      float v = colv ? fmaxf(cv[r] + biv, 0.f) : 0.f;
      Qhp[o] = (half_t)q;
      Khp[o] = (half_t)k;
      Vhp[o] = (half_t)v;
    }
  }
}

// ---------------- Sparsity measure (Q row hoisted, 8-half vector dots) ----------------
__global__ __launch_bounds__(256)
void measure_kernel(const half_t* __restrict__ Qhp, const half_t* __restrict__ Khp,
                    float* __restrict__ Mm) {
  int t = blockIdx.x * 256 + threadIdx.x;
  if (t >= BH * LL) return;
  int bh = t / LL;
  int q  = t - bh * LL;
  const half_t* Qrow  = Qhp + (size_t)(bh * LLP + q) * DP;
  const half_t* Kbase = Khp + (size_t)bh * SBH;
  H8 qa[DD / 8];
#pragma unroll
  for (int d8 = 0; d8 < DD / 8; ++d8) qa[d8].q = *(const uint4*)(Qrow + d8 * 8);
  float mx = neg_inf(), sm = 0.f;
  for (int s = 0; s < SAMPK; ++s) {
    unsigned idx = (1103515245u * (unsigned)(q * SAMPK + s) + 12345u) % (unsigned)LL;
    const half_t* Krow = Kbase + (size_t)idx * DP;
    float acc = 0.f;
#pragma unroll
    for (int d8 = 0; d8 < DD / 8; ++d8) {
      H8 b;
      b.q = *(const uint4*)(Krow + d8 * 8);
#pragma unroll
      for (int j = 0; j < 8; ++j) acc += (float)qa[d8].h[j] * (float)b.h[j];
    }
    mx = fmaxf(mx, acc);
    sm += acc;
  }
  Mm[t] = mx - sm / (float)LL;
}

// ---------------- top-70 per (b,h) ----------------
__global__ __launch_bounds__(256)
void topk_kernel(const float* __restrict__ Mm, int* __restrict__ Mtop) {
  __shared__ float v[LLP];
  __shared__ float rv[256];
  __shared__ int   ri[256];
  const int bh = blockIdx.x, tid = threadIdx.x;
  for (int i = tid; i < LLP; i += 256) v[i] = (i < LL) ? Mm[bh * LL + i] : neg_inf();
  __syncthreads();
  for (int it = 0; it < NTOP; ++it) {
    float best = neg_inf(); int bi = 0;
    for (int i = tid; i < LLP; i += 256)
      if (v[i] > best) { best = v[i]; bi = i; }
    rv[tid] = best; ri[tid] = bi;
    __syncthreads();
    for (int s = 128; s > 0; s >>= 1) {
      if (tid < s && rv[tid + s] > rv[tid]) { rv[tid] = rv[tid + s]; ri[tid] = ri[tid + s]; }
      __syncthreads();
    }
    if (tid == 0) { Mtop[bh * NTOP + it] = ri[0]; v[ri[0]] = neg_inf(); }
    __syncthreads();
  }
}

// ---------------- context = cumsum(V) written as (B, N, T, D) ----------------
__global__ void cumsum_kernel(const half_t* __restrict__ Vhp, float* __restrict__ out) {
  const int bh = blockIdx.x;
  const int b = bh / HH, h = bh - b * HH;
  const int d = threadIdx.x;
  if (d >= DD) return;
  const half_t* src = Vhp + (size_t)bh * SBH + d;
  float run = 0.f;
  for (int n = 0; n < LL; ++n) {
    run += (float)src[(size_t)n * DP];
    out[(((size_t)b * LL + n) * HH + h) * DD + d] = run;
  }
}

// ---------------- V transpose: Vt[bh][160][512] (B layout for attn@V) ----------------
__global__ __launch_bounds__(256)
void vt_kernel(const half_t* __restrict__ Vhp, half_t* __restrict__ Vt) {
  int id = blockIdx.x * 256 + threadIdx.x;          // one dword (pair along n)
  if (id >= BH * DP * (LLP / 2)) return;
  int bh = id / (DP * (LLP / 2));
  int rem = id - bh * (DP * (LLP / 2));
  int e = rem / (LLP / 2);
  int pd = rem - e * (LLP / 2);
  int n = pd * 2;
  const half_t* src = Vhp + (size_t)bh * SBH;
  bool ok0 = (n < LL), ok1 = (n + 1 < LL);
  H2 t;
  t.h[0] = ok0 ? src[(size_t)n * DP + e] : (_Float16)0.f;
  t.h[1] = ok1 ? src[(size_t)(n + 1) * DP + e] : (_Float16)0.f;
  ((unsigned*)Vt)[id] = t.u;
}

// ---------------- scores -> mask -> softmax -> attn@V -> scatter ----------------
__global__ __launch_bounds__(32)
void sparse_attn_kernel(const half_t* __restrict__ Qhp, const half_t* __restrict__ Khp,
                        const half_t* __restrict__ Vt, const int* __restrict__ Mtop,
                        float* __restrict__ out) {
  __shared__ float  sc[16][LLP];       // 32 KB scores (f32 softmax)
  __shared__ half_t at[16][LLP];       // 16 KB probs (f16 WMMA A)
  __shared__ half_t sQ[16 * DP];       // 5 KB gathered Q_reduce rows
  __shared__ int qi[16];
  __shared__ int mk[16];

  const int lane = threadIdx.x;
  const int bh = blockIdx.x;
  const int b = bh / HH, h = bh - b * HH;
  const int u0 = blockIdx.y * 16;
  const float scale = 0.081111071056538127f;   // 1/sqrt(152)

  const half_t* Qb = Qhp + (size_t)bh * SBH;
  const half_t* Kb = Khp + (size_t)bh * SBH;
  const half_t* Vb = Vt  + (size_t)bh * SBH;

  if (lane < 16) {
    int u = u0 + lane;
    int idx = (u < NTOP) ? Mtop[bh * NTOP + u] : 0;
    qi[lane] = idx;
    mk[lane] = (u < NTOP) ? idx : (LL - 1);
  }
  __syncthreads();

  // Gather 16 Q rows into LDS via per-lane async global->LDS DMA (ASYNCcnt-tracked).
  // LDS byte offset = low 32 bits of generic pointer (LDS aperture: addr[31:0]).
  {
    unsigned lds_base = (unsigned)(unsigned long long)(&sQ[0]);
    for (int i = lane; i < 16 * (DP / 2); i += 32) {
      int r = i / (DP / 2), pr = i - r * (DP / 2);
      unsigned long long ga = (unsigned long long)(Qb + (size_t)qi[r] * DP + pr * 2);
      unsigned la = lds_base + (unsigned)i * 4u;
      asm volatile("global_load_async_to_lds_b32 %0, %1, off"
                   :: "v"(la), "v"(ga) : "memory");
    }
    asm volatile("s_wait_asynccnt 0x0" ::: "memory");
  }
  __syncthreads();

  // ---- scores = (Q_reduce @ K^T) * scale, masked; Khp row-major IS the B layout ----
  for (int nt = 0; nt < LLP / 16; ++nt) {
    const int n0 = nt * 16;
    if (nt + 1 < LLP / 16) __builtin_prefetch(Kb + (n0 + 16) * DP, 0, 1);
    v8f c = {};
#pragma unroll
    for (int ks = 0; ks < KSTEPS_D; ++ks) {
      v16h a  = frag_a(sQ, DP, lane & 15, ks * 32, lane);
      v16h bm = frag_bt(Kb + n0 * DP, DP, ks * 32, lane);
      c = WMMA_F16(a, bm, c);
    }
    const int n = n0 + (lane & 15);
    const int rb = (lane >> 4) << 3;
#pragma unroll
    for (int r = 0; r < 8; ++r) {
      int row = rb + r;
      float s = c[r] * scale;
      if (n >= LL || n > mk[row]) s = neg_inf();
      sc[row][n] = s;
    }
  }
  __syncthreads();

  // ---- softmax along L (wave32 shfl reductions), emit f16 probs ----
  for (int r = 0; r < 16; ++r) {
    float mx = neg_inf();
    for (int c = lane; c < LLP; c += 32) mx = fmaxf(mx, sc[r][c]);
#pragma unroll
    for (int o = 16; o > 0; o >>= 1) mx = fmaxf(mx, __shfl_xor(mx, o, 32));
    float sm = 0.f;
    for (int c = lane; c < LLP; c += 32) {
      float e = __expf(sc[r][c] - mx);
      sc[r][c] = e;
      sm += e;
    }
#pragma unroll
    for (int o = 16; o > 0; o >>= 1) sm += __shfl_xor(sm, o, 32);
    float inv = 1.f / sm;
    for (int c = lane; c < LLP; c += 32) at[r][c] = (half_t)(sc[r][c] * inv);
  }
  __syncthreads();

  // ---- upd = attn @ V via Vt panels; scatter rows to out[b, M_top[u], h, :] ----
  for (int et = 0; et < NTILES_D; ++et) {
    const int e0 = et * 16;
    if (et + 1 < NTILES_D) __builtin_prefetch(Vb + (e0 + 16) * LLP, 0, 1);
    v8f c = {};
#pragma unroll
    for (int ks = 0; ks < LLP / 32; ++ks) {
      v16h a  = frag_a(&at[0][0], LLP, lane & 15, ks * 32, lane);
      v16h bm = frag_bt(Vb + e0 * LLP, LLP, ks * 32, lane);
      c = WMMA_F16(a, bm, c);
    }
    const int e = e0 + (lane & 15);
    const bool ev = e < DD;
    const int rb = (lane >> 4) << 3;
#pragma unroll
    for (int r = 0; r < 8; ++r) {
      int u = u0 + rb + r;
      if (ev && u < NTOP) {
        int qrow = qi[rb + r];
        out[(((size_t)b * LL + qrow) * HH + h) * DD + e] = c[r];
      }
    }
  }
}

// ---------------- Launch ----------------
extern "C" void kernel_launch(void* const* d_in, const int* in_sizes, int n_in,
                              void* d_out, int out_size, void* d_ws, size_t ws_size,
                              hipStream_t stream) {
  const float* X  = (const float*)d_in[0];
  const float* Wq = (const float*)d_in[1];
  const float* bq = (const float*)d_in[2];
  const float* Wk = (const float*)d_in[3];
  const float* bk = (const float*)d_in[4];
  const float* Wv = (const float*)d_in[5];
  const float* bv = (const float*)d_in[6];
  float* out = (float*)d_out;

  // Workspace carve-up (~79 MB total, all 16B+ aligned)
  char* w = (char*)d_ws;
  half_t* Xhp = (half_t*)w;  w += (size_t)BH * SBH * sizeof(half_t);   // 15.7 MB
  half_t* Qhp = (half_t*)w;  w += (size_t)BH * SBH * sizeof(half_t);
  half_t* Khp = (half_t*)w;  w += (size_t)BH * SBH * sizeof(half_t);
  half_t* Vhp = (half_t*)w;  w += (size_t)BH * SBH * sizeof(half_t);
  half_t* Vt  = (half_t*)w;  w += (size_t)BH * SBH * sizeof(half_t);
  half_t* Wp  = (half_t*)w;  w += (size_t)3 * DP * DP * sizeof(half_t);
  float*  Mm  = (float*)w;   w += (size_t)BH * LL * sizeof(float);
  int*  Mtop  = (int*)w;

  int nxd = BH * LLP * (DP / 2);
  prep_x_kernel<<<(nxd + 255) / 256, 256, 0, stream>>>(X, Xhp);
  int nwd = 3 * DP * (DP / 2);
  prep_w_kernel<<<(nwd + 255) / 256, 256, 0, stream>>>(Wq, Wk, Wv, Wp);

  qkv_kernel<<<BH * LLP / 16, 32, 0, stream>>>(Xhp, Wp, bq, bk, bv, Qhp, Khp, Vhp);

  measure_kernel<<<(BH * LL + 255) / 256, 256, 0, stream>>>(Qhp, Khp, Mm);
  topk_kernel<<<BH, 256, 0, stream>>>(Mm, Mtop);
  cumsum_kernel<<<BH, 160, 0, stream>>>(Vhp, out);

  int nvd = BH * DP * (LLP / 2);
  vt_kernel<<<(nvd + 255) / 256, 256, 0, stream>>>(Vhp, Vt);

  sparse_attn_kernel<<<dim3(BH, (NTOP + 15) / 16), 32, 0, stream>>>(Qhp, Khp, Vt, Mtop, out);
}
